// Head_34643206209675
// MI455X (gfx1250) — matrix operand order repaired and checked by reading, not compiled, not run on hardware
//
#include <hip/hip_runtime.h>
#include <hip/hip_bf16.h>

typedef __attribute__((ext_vector_type(16))) _Float16 v16h;
typedef __attribute__((ext_vector_type(8)))  float    v8f;
typedef __attribute__((ext_vector_type(2)))  __fp16   v2fp16;   // cvt_pkrtz return type

#define B_   8
#define T_   2048
#define C_   768
#define D_   64
#define NT_  192                       // 3 * D (q | k | v fused projection)
#define NEG_FILL (-99999999999999.0f)  // exact constant from the source

union V16 { v16h v; uint4 q[2]; int i[8]; };
union PK  { v2fp16 h; int i; };

__device__ __forceinline__ int pack2(float a, float b) {
    PK p; p.h = __builtin_amdgcn_cvt_pkrtz(a, b); return p.i;
}

// ---------------- kernel 1: x fp32 -> f16 (one streaming pass) ----------------
__global__ void __launch_bounds__(256)
cvt_x_f16(const float* __restrict__ x, _Float16* __restrict__ x16, int n) {
    int i = (blockIdx.x * 256 + threadIdx.x) * 4;
    if (i + 3 < n) {
        float4 f = *(const float4*)(x + i);
        int2 o;
        o.x = pack2(f.x, f.y);
        o.y = pack2(f.z, f.w);
        *(int2*)(x16 + i) = o;
    }
}

// ------------- kernel 2: Wt[n][c] = concat(Wq,Wk,Wv)[c][n] in f16 -------------
__global__ void __launch_bounds__(256)
build_wt(const float* __restrict__ Wk, const float* __restrict__ Wq,
         const float* __restrict__ Wv, _Float16* __restrict__ Wt) {
    int tid = blockIdx.x * 256 + threadIdx.x;
    if (tid >= NT_ * C_) return;
    int n = tid / C_, c = tid % C_;
    float v;
    if (n < 64)       v = Wq[c * 64 + n];
    else if (n < 128) v = Wk[c * 64 + (n - 64)];
    else              v = Wv[c * 64 + (n - 128)];
    Wt[n * C_ + c] = (_Float16)v;
}

// --- kernel 3: fused qkv projection GEMM [16384x768]x[768x192], WMMA f16 ---
// one 16x16 output tile per wave; q,k stored [B,T,64]; v stored transposed [B,64,T]
__global__ void __launch_bounds__(128)
proj_qkv(const _Float16* __restrict__ x16, const _Float16* __restrict__ Wt,
         _Float16* __restrict__ qf, _Float16* __restrict__ kf,
         _Float16* __restrict__ vT) {
    const int lane = threadIdx.x & 31;
    const int w    = blockIdx.x * 4 + (threadIdx.x >> 5);
    const int mt   = w / (NT_ / 16), nt = w % (NT_ / 16);
    const int m0   = mt * 16, n0 = nt * 16;
    const int col  = lane & 15, h = lane >> 4;

    const _Float16* arow = x16 + (size_t)(m0 + col) * C_;
    const _Float16* brow = Wt  + (size_t)(n0 + col) * C_;

    v8f acc = {};
    #pragma unroll 4
    for (int kk = 0; kk < C_; kk += 32) {
        V16 a, b;
        // A (16-bit 16x32 layout): K = 8h..8h+7 and 8h+16..8h+23
        a.q[0] = *(const uint4*)(arow + kk + 8 * h);
        a.q[1] = *(const uint4*)(arow + kk + 8 * h + 16);
        // B (32x16 layout): K = 8h..8h+7 and 8h+16..8h+23 (same as A)
        b.q[0] = *(const uint4*)(brow + kk + 8 * h);
        b.q[1] = *(const uint4*)(brow + kk + 8 * h + 16);
        acc = __builtin_amdgcn_wmma_f32_16x16x32_f16(false, a.v, false, b.v,
                                                     (short)0, acc, false, false);
    }
    // C/D layout: value(M = r + 8h, N = col)
    if (n0 < 128) {
        _Float16* dst = (n0 < 64) ? (qf + n0 + col) : (kf + (n0 - 64) + col);
        #pragma unroll
        for (int r = 0; r < 8; ++r)
            dst[(size_t)(m0 + r + 8 * h) * D_] = (_Float16)acc[r];
    } else {
        int d  = n0 - 128 + col;
        int b_ = m0 / T_;
        int t0 = (m0 % T_) + 8 * h;                 // r=0..7 -> consecutive t
        uint4 o;
        o.x = pack2(acc[0], acc[1]);
        o.y = pack2(acc[2], acc[3]);
        o.z = pack2(acc[4], acc[5]);
        o.w = pack2(acc[6], acc[7]);
        *(uint4*)(vT + (size_t)b_ * D_ * T_ + (size_t)d * T_ + t0) = o;
    }
}

// ----- kernel 4: streaming attention, faithful multiplicative mask + softmax ----
// per wave: 16 query rows; loop all 2048 keys in chunks of 32.
// Computes S^T = K*Q^T so per-row softmax stats are a single shfl_xor(16).
__global__ void __launch_bounds__(128)
attn(const _Float16* __restrict__ qf, const _Float16* __restrict__ kf,
     const _Float16* __restrict__ vT, float* __restrict__ out) {
    const int lane = threadIdx.x & 31;
    const int w    = blockIdx.x * 4 + (threadIdx.x >> 5);
    const int b    = w >> 7;                 // / (T/16)
    const int t0   = (w & 127) * 16;
    const int col  = lane & 15, h = lane >> 4;
    const int t_g  = t0 + col;               // this lane's query row

    const _Float16* qb = qf + (size_t)b * T_ * D_;
    const _Float16* kb = kf + (size_t)b * T_ * D_;
    const _Float16* vb = vT + (size_t)b * D_ * T_;

    // hoist Q^T B-operands (two 32-wide d chunks), K = 8h..8h+7 and 8h+16..8h+23
    V16 qB[2];
    {
        const _Float16* qrow = qb + (size_t)t_g * D_;
        qB[0].q[0] = *(const uint4*)(qrow + 8 * h);
        qB[0].q[1] = *(const uint4*)(qrow + 8 * h + 16);
        qB[1].q[0] = *(const uint4*)(qrow + 32 + 8 * h);
        qB[1].q[1] = *(const uint4*)(qrow + 32 + 8 * h + 16);
    }

    v8f acc[4] = {{}, {}, {}, {}};           // out^T tiles, d = 16n + M
    float m_run = -3.0e38f, l_run = 0.0f;
    const float MASK_NEG = NEG_FILL * 0.125f;

    for (int it = 0; it < T_ / 32; ++it) {
        const int sbase = it * 32;
        if (it + 1 < T_ / 32)                // warm L2/L0 for next K chunk
            __builtin_prefetch(kb + (size_t)(sbase + 32 + col) * D_, 0, 0);

        // ---- S^T tiles: A = K rows (M=s), B = Q^T ----
        v8f st[2];
        #pragma unroll
        for (int tile = 0; tile < 2; ++tile) {
            const _Float16* krow = kb + (size_t)(sbase + 16 * tile + col) * D_;
            V16 a0, a1;
            a0.q[0] = *(const uint4*)(krow + 8 * h);
            a0.q[1] = *(const uint4*)(krow + 8 * h + 16);
            a1.q[0] = *(const uint4*)(krow + 32 + 8 * h);
            a1.q[1] = *(const uint4*)(krow + 32 + 8 * h + 16);
            v8f s = {};
            s = __builtin_amdgcn_wmma_f32_16x16x32_f16(false, a0.v, false, qB[0].v,
                                                       (short)0, s, false, false);
            s = __builtin_amdgcn_wmma_f32_16x16x32_f16(false, a1.v, false, qB[1].v,
                                                       (short)0, s, false, false);
            st[tile] = s;
        }

        // ---- faithful mask: logit = raw * (s<=t ? 1 : NEG_FILL) * D^-0.5 ----
        float lg[2][8];
        float mx = -3.0e38f;
        #pragma unroll
        for (int tile = 0; tile < 2; ++tile)
            #pragma unroll
            for (int r = 0; r < 8; ++r) {
                int s = sbase + 16 * tile + r + 8 * h;
                float v = st[tile][r] * ((s <= t_g) ? 0.125f : MASK_NEG);
                lg[tile][r] = v;
                mx = fmaxf(mx, v);
            }
        mx = fmaxf(mx, __shfl_xor(mx, 16, 32));     // row lives in lanes {l, l^16}
        float m_new = fmaxf(m_run, mx);
        float f = __expf(m_run - m_new);
        m_run = m_new;

        float p[2][8];
        float ls = 0.0f;
        #pragma unroll
        for (int tile = 0; tile < 2; ++tile)
            #pragma unroll
            for (int r = 0; r < 8; ++r) {
                float e = __expf(lg[tile][r] - m_new);
                p[tile][r] = e;
                ls += e;
            }
        ls += __shfl_xor(ls, 16, 32);
        l_run = l_run * f + ls;

        #pragma unroll
        for (int n = 0; n < 4; ++n)
            #pragma unroll
            for (int r = 0; r < 8; ++r)
                acc[n][r] *= f;

        // ---- C-layout P^T -> B-layout (static regs, lane^16 exchange only) ----
        int t0p[4], t1p[4];
        #pragma unroll
        for (int j = 0; j < 4; ++j) {
            t0p[j] = pack2(p[0][2 * j], p[0][2 * j + 1]);
            t1p[j] = pack2(p[1][2 * j], p[1][2 * j + 1]);
        }
        V16 pB;
        #pragma unroll
        for (int j = 0; j < 4; ++j) {
            int o0 = __shfl_xor(t0p[j], 16, 32);
            int o1 = __shfl_xor(t1p[j], 16, 32);
            pB.i[j]     = (h == 0) ? t0p[j] : o1;   // K = 16h + 2j, 2j+1
            pB.i[4 + j] = (h == 0) ? o0 : t1p[j];   // K = 16h + 2j+8, 2j+9
        }

        // ---- out^T += V^T * P^T ----
        #pragma unroll
        for (int n = 0; n < 4; ++n) {
            const _Float16* vrow = vb + (size_t)(16 * n + col) * T_ + sbase;
            V16 a;
            a.q[0] = *(const uint4*)(vrow + 16 * h);
            a.q[1] = *(const uint4*)(vrow + 16 * h + 8);
            acc[n] = __builtin_amdgcn_wmma_f32_16x16x32_f16(false, a.v, false, pB.v,
                                                            (short)0, acc[n], false, false);
        }
    }

    // ---- normalize and store: lane owns row t_g, 8 consecutive d per acc ----
    const float inv = 1.0f / l_run;
    float* orow = out + ((size_t)(b * T_ + t_g) * D_);
    #pragma unroll
    for (int n = 0; n < 4; ++n) {
        float4 lo, hi;
        lo.x = acc[n][0] * inv; lo.y = acc[n][1] * inv;
        lo.z = acc[n][2] * inv; lo.w = acc[n][3] * inv;
        hi.x = acc[n][4] * inv; hi.y = acc[n][5] * inv;
        hi.z = acc[n][6] * inv; hi.w = acc[n][7] * inv;
        float* dst = orow + 16 * n + 8 * h;
        *(float4*)(dst)     = lo;
        *(float4*)(dst + 4) = hi;
    }
}

extern "C" void kernel_launch(void* const* d_in, const int* in_sizes, int n_in,
                              void* d_out, int out_size, void* d_ws, size_t ws_size,
                              hipStream_t stream) {
    const float* x  = (const float*)d_in[0];
    const float* Wk = (const float*)d_in[1];
    const float* Wq = (const float*)d_in[2];
    const float* Wv = (const float*)d_in[3];
    float* out = (float*)d_out;

    // workspace layout (all f16): x16 | Wt | q | k | vT  (~31.8 MB total)
    _Float16* x16 = (_Float16*)d_ws;
    _Float16* Wt  = x16 + (size_t)B_ * T_ * C_;
    _Float16* qf  = Wt  + (size_t)NT_ * C_;
    _Float16* kf  = qf  + (size_t)B_ * T_ * D_;
    _Float16* vT  = kf  + (size_t)B_ * T_ * D_;

    const int nx = B_ * T_ * C_;
    hipLaunchKernelGGL(cvt_x_f16, dim3(nx / 4 / 256), dim3(256), 0, stream, x, x16, nx);
    hipLaunchKernelGGL(build_wt, dim3((NT_ * C_ + 255) / 256), dim3(256), 0, stream,
                       Wk, Wq, Wv, Wt);
    // 1024 M-tiles * 12 N-tiles = 12288 waves, 4 waves/block
    hipLaunchKernelGGL(proj_qkv, dim3(12288 / 4), dim3(128), 0, stream,
                       x16, Wt, qf, kf, vT);
    // 8 batches * 128 row-tiles = 1024 waves, 4 waves/block
    hipLaunchKernelGGL(attn, dim3(1024 / 4), dim3(128), 0, stream, qf, kf, vT, out);
}
